// ConvBlock_26929444946693
// MI455X (gfx1250) — matrix operand order repaired and checked by reading, not compiled, not run on hardware
//
#include <hip/hip_runtime.h>
#include <hip/hip_bf16.h>
#include <math.h>

typedef __attribute__((ext_vector_type(16))) _Float16 v16h;
typedef __attribute__((ext_vector_type(8)))  _Float16 h8;
typedef __attribute__((ext_vector_type(8)))  float    v8f;

#define NNODE  50000
#define NEDGE  800000
#define CH     64
#define NKERN  64
#define NPAIR  (NEDGE * 8)
#define NCLUST (16 * 512)
#define GPB    128   // workgroups per kernel-bin in the spline GEMM

// ---------------- helpers ----------------

// degree-1 open B-spline basis over 3 dims, K=4 knots -> 8 corners
__device__ inline void spline_basis(const float* __restrict__ attr, float* w8, int* k8) {
    float tt[3], lo[3];
#pragma unroll
    for (int d = 0; d < 3; ++d) {
        float fr = attr[d] * 3.0f;              // K-1 = 3
        float l  = floorf(fr);
        l        = fminf(fmaxf(l, 0.0f), 2.0f); // clip to [0, K-2]
        tt[d]    = fr - l;
        lo[d]    = l;
    }
#pragma unroll
    for (int b = 0; b < 8; ++b) {
        float w = 1.0f;
        int   kk = 0;
#pragma unroll
        for (int d = 0; d < 3; ++d) {
            int bit = (b >> d) & 1;
            w  *= bit ? tt[d] : (1.0f - tt[d]);
            int stride = (d == 0) ? 1 : (d == 1) ? 4 : 16;
            kk += ((int)lo[d] + bit) * stride;
        }
        w8[b] = w;
        k8[b] = kk;
    }
}

// A fragment (16x32 f16): lane half hi selects K base 0/8; halves j<8 -> K=kb+j,
// j>=8 -> K=kb+8+j (ISA 7.12.2 16-bit A layout). Scaled by per-row basis.
__device__ inline v16h load_a_frag(const _Float16* __restrict__ rowp, int kk, int hi, float s) {
    int kbase = kk + (hi ? 8 : 0);
    h8 a0 = *(const h8*)(rowp + kbase);        // K = kbase .. kbase+7
    h8 a1 = *(const h8*)(rowp + kbase + 16);   // K = kbase+16 .. kbase+23
    _Float16 hs = (_Float16)s;
    v16h r;
#pragma unroll
    for (int j = 0; j < 8; ++j) { r[j] = a0[j] * hs; r[j + 8] = a1[j] * hs; }
    return r;
}

// B fragment (32x16 f16), W row-major [K=64][N=64]: same K order as A fragment:
// kbase = kk + 8*hi; j<8 -> K=kbase+j, j>=8 -> K=kbase+8+j.
__device__ inline v16h load_b_frag(const _Float16* __restrict__ w, int kk, int t, int lane) {
    int n     = t * 16 + (lane & 15);
    int kbase = kk + ((lane >= 16) ? 8 : 0);
    v16h r;
#pragma unroll
    for (int j = 0; j < 16; ++j) r[j] = w[(size_t)(kbase + j + (j >= 8 ? 8 : 0)) * CH + n];
    return r;
}

__device__ inline unsigned flipf(float f) {
    unsigned b = __float_as_uint(f);
    return b ^ ((b & 0x80000000u) ? 0xFFFFFFFFu : 0x80000000u);
}
__device__ inline float unflipf(unsigned u) {
    unsigned b = u ^ ((u & 0x80000000u) ? 0x80000000u : 0xFFFFFFFFu);
    return __uint_as_float(b);
}

// ---------------- kernels ----------------

// convert to f16 + zero/init all accumulators (grid-stride free: covers max range)
__global__ __launch_bounds__(256) void prep_kernel(
    const float* __restrict__ x, const float* __restrict__ w, const float* __restrict__ root,
    _Float16* xh, _Float16* wh, _Float16* rooth,
    float* hb, float* cnt, float* stats, int* bin_count, unsigned* keys) {
    int i = blockIdx.x * 256 + threadIdx.x;
    if (i < NNODE * CH)      { xh[i] = (_Float16)x[i]; hb[i] = 0.0f; }
    if (i < NKERN * CH * CH) { wh[i] = (_Float16)w[i]; }
    if (i < CH * CH)         { rooth[i] = (_Float16)root[i]; }
    if (i < NNODE)           { cnt[i] = 0.0f; }
    if (i < 128)             { stats[i] = 0.0f; }
    if (i < 64)              { bin_count[i] = 0; }
    if (i < NCLUST * CH)     { keys[i] = 0u; }
}

// histogram (edge,corner) pairs into 64 kernel bins; count in-degree
__global__ __launch_bounds__(256) void count_kernel(
    const int* __restrict__ ei, const float* __restrict__ ea,
    int* bin_count, float* cnt) {
    __shared__ int hist[64];
    int t = threadIdx.x;
    if (t < 64) hist[t] = 0;
    __syncthreads();
    int e = blockIdx.x * 256 + t;
    if (e < NEDGE) {
        float w8[8]; int k8[8];
        spline_basis(ea + (size_t)e * 3, w8, k8);
#pragma unroll
        for (int b = 0; b < 8; ++b) atomicAdd(&hist[k8[b]], 1);
        atomicAdd(&cnt[ei[NEDGE + e]], 1.0f);
    }
    __syncthreads();
    if (t < 64) atomicAdd(&bin_count[t], hist[t]);
}

__global__ void scan_kernel(const int* __restrict__ bin_count, int* bin_base, int* bin_cursor) {
    if (threadIdx.x == 0 && blockIdx.x == 0) {
        int acc = 0;
        for (int k = 0; k < 64; ++k) { bin_base[k] = acc; bin_cursor[k] = acc; acc += bin_count[k]; }
        bin_base[64] = acc;
    }
}

// scatter pair records {src,dst,basis} into kernel-sorted order
__global__ __launch_bounds__(256) void scatter_kernel(
    const int* __restrict__ ei, const float* __restrict__ ea,
    int* bin_cursor, int* rec_src, int* rec_dst, float* rec_basis) {
    int e = blockIdx.x * 256 + threadIdx.x;
    if (e >= NEDGE) return;
    int src = ei[e], dst = ei[NEDGE + e];
    float w8[8]; int k8[8];
    spline_basis(ea + (size_t)e * 3, w8, k8);
#pragma unroll
    for (int b = 0; b < 8; ++b) {
        int p = atomicAdd(&bin_cursor[k8[b]], 1);
        rec_src[p] = src; rec_dst[p] = dst; rec_basis[p] = w8[b];
    }
}

// per-bin WMMA GEMM: (basis * x[src])[16 x 64] @ W_k[64 x 64] -> atomic scatter to agg[dst]
__global__ __launch_bounds__(32) void spline_gemm_kernel(
    const int* __restrict__ rec_src, const int* __restrict__ rec_dst,
    const float* __restrict__ rec_basis, const int* __restrict__ bin_base,
    const _Float16* __restrict__ xh, const _Float16* __restrict__ wh,
    float* __restrict__ agg) {
    int lane = threadIdx.x;
    int k    = blockIdx.x / GPB;
    int g    = blockIdx.x % GPB;
    int base   = bin_base[k];
    int count  = bin_base[k + 1] - base;
    int ntiles = (count + 15) >> 4;
    const _Float16* wk = wh + (size_t)k * CH * CH;

    v16h B[2][4];   // kept in registers across all tiles of this bin
#pragma unroll
    for (int kk = 0; kk < 2; ++kk)
#pragma unroll
        for (int t = 0; t < 4; ++t) B[kk][t] = load_b_frag(wk, kk * 32, t, lane);

    int hi  = (lane >= 16) ? 1 : 0;
    int m   = lane & 15;
    int nlo = lane & 15;

    for (int tile = g; tile < ntiles; tile += GPB) {
        int  pi    = tile * 16 + m;
        bool valid = pi < count;
        int  p     = base + (valid ? pi : 0);
        int  src   = valid ? rec_src[p]   : 0;
        int  dstv  = valid ? rec_dst[p]   : -1;
        float bas  = valid ? rec_basis[p] : 0.0f;

        const _Float16* rowp = xh + (size_t)src * CH;
        v16h A0 = load_a_frag(rowp, 0,  hi, bas);
        v16h A1 = load_a_frag(rowp, 32, hi, bas);

        v8f c[4] = {};
#pragma unroll
        for (int t = 0; t < 4; ++t) {
            c[t] = __builtin_amdgcn_wmma_f32_16x16x32_f16(false, A0, false, B[0][t], (short)0, c[t], false, false);
            c[t] = __builtin_amdgcn_wmma_f32_16x16x32_f16(false, A1, false, B[1][t], (short)0, c[t], false, false);
        }
        // scatter: C row r on this lane is pair m = r + 8*hi; its dst lives on lane m
#pragma unroll
        for (int r = 0; r < 8; ++r) {
            int mr = r + 8 * hi;
            int d  = __shfl(dstv, mr, 32);
            if (d >= 0) {
#pragma unroll
                for (int t = 0; t < 4; ++t)
                    atomicAdd(&agg[(size_t)d * CH + t * 16 + nlo], c[t][r]);
            }
        }
    }
}

// h = agg/max(cnt,1) + x@root + bias ; ELU ; accumulate batch sum/sumsq
__global__ __launch_bounds__(32) void root_elu_stats_kernel(
    const _Float16* __restrict__ xh, const _Float16* __restrict__ rooth,
    const float* __restrict__ cnt, const float* __restrict__ bias,
    float* __restrict__ hb, float* __restrict__ stats) {
    int lane = threadIdx.x;
    int tile = blockIdx.x;
    int hi   = (lane >= 16) ? 1 : 0;
    int m    = lane & 15;
    int nlo  = lane & 15;

    int  node_a = tile * 16 + m;
    bool va     = node_a < NNODE;
    const _Float16* rowp = xh + (size_t)(va ? node_a : 0) * CH;
    float sc = va ? 1.0f : 0.0f;
    v16h A0 = load_a_frag(rowp, 0,  hi, sc);
    v16h A1 = load_a_frag(rowp, 32, hi, sc);

    v8f c[4] = {};
#pragma unroll
    for (int t = 0; t < 4; ++t) {
        v16h b0 = load_b_frag(rooth, 0,  t, lane);
        v16h b1 = load_b_frag(rooth, 32, t, lane);
        c[t] = __builtin_amdgcn_wmma_f32_16x16x32_f16(false, A0, false, b0, (short)0, c[t], false, false);
        c[t] = __builtin_amdgcn_wmma_f32_16x16x32_f16(false, A1, false, b1, (short)0, c[t], false, false);
    }

    float sum[4] = {0, 0, 0, 0}, sq[4] = {0, 0, 0, 0};
#pragma unroll
    for (int t = 0; t < 4; ++t) {
        int n = t * 16 + nlo;
        float bi = bias[n];
#pragma unroll
        for (int r = 0; r < 8; ++r) {
            int node = tile * 16 + r + 8 * hi;
            if (node < NNODE) {
                size_t idx = (size_t)node * CH + n;
                float deg  = fmaxf(cnt[node], 1.0f);
                float v    = c[t][r] + hb[idx] / deg + bi;
                v = (v > 0.0f) ? v : (expf(v) - 1.0f);   // ELU
                hb[idx] = v;
                sum[t] += v;
                sq[t]  += v * v;
            }
        }
    }
#pragma unroll
    for (int t = 0; t < 4; ++t) {
        float s = sum[t] + __shfl_xor(sum[t], 16, 32);
        float q = sq[t]  + __shfl_xor(sq[t],  16, 32);
        if (lane < 16) {
            atomicAdd(&stats[t * 16 + lane], s);
            atomicAdd(&stats[64 + t * 16 + lane], q);
        }
    }
}

// BatchNorm + voxel-grid max pool (monotonic uint atomicMax)
__global__ __launch_bounds__(256) void bn_pool_kernel(
    const float* __restrict__ hb, const float* __restrict__ stats,
    const float* __restrict__ gamma, const float* __restrict__ beta,
    const float* __restrict__ pos, const int* __restrict__ batch,
    unsigned* __restrict__ keys) {
    int i = blockIdx.x * 256 + threadIdx.x;
    if (i >= NNODE * CH) return;
    int node = i >> 6, d = i & 63;
    float mean = stats[d] * (1.0f / NNODE);
    float var  = stats[64 + d] * (1.0f / NNODE) - mean * mean;
    float v = (hb[i] - mean) * rsqrtf(var + 1e-5f) * gamma[d] + beta[d];
    int v0 = min(max((int)floorf(pos[node * 3 + 0] * (1.0f / 32.0f)), 0), 7);
    int v1 = min(max((int)floorf(pos[node * 3 + 1] * (1.0f / 32.0f)), 0), 7);
    int v2 = min(max((int)floorf(pos[node * 3 + 2] * (1.0f / 32.0f)), 0), 7);
    int cluster = batch[node] * 512 + v0 * 64 + v1 * 8 + v2;
    atomicMax(&keys[(size_t)cluster * CH + d], flipf(v));
}

__global__ __launch_bounds__(256) void pool_finalize_kernel(
    const unsigned* __restrict__ keys, float* __restrict__ out) {
    int i = blockIdx.x * 256 + threadIdx.x;
    if (i >= NCLUST * CH) return;
    unsigned u = keys[i];
    out[i] = (u == 0u) ? 0.0f : unflipf(u);  // empty voxel -> 0
}

// ---------------- host launch ----------------

extern "C" void kernel_launch(void* const* d_in, const int* in_sizes, int n_in,
                              void* d_out, int out_size, void* d_ws, size_t ws_size,
                              hipStream_t stream) {
    const float* x      = (const float*)d_in[0];
    const int*   ei     = (const int*)d_in[1];
    const float* ea     = (const float*)d_in[2];
    const float* pos    = (const float*)d_in[3];
    const int*   batch  = (const int*)d_in[4];
    const float* weight = (const float*)d_in[5];
    const float* root   = (const float*)d_in[6];
    const float* bias   = (const float*)d_in[7];
    const float* gamma  = (const float*)d_in[8];
    const float* beta   = (const float*)d_in[9];
    float* out = (float*)d_out;

    char* p = (char*)d_ws;
    auto take = [&](size_t bytes) -> void* {
        void* r = (void*)p;
        p += (bytes + 255) & ~(size_t)255;
        return r;
    };
    _Float16* xh        = (_Float16*)take((size_t)NNODE * CH * 2);
    _Float16* wh        = (_Float16*)take((size_t)NKERN * CH * CH * 2);
    _Float16* rooth     = (_Float16*)take((size_t)CH * CH * 2);
    float*    hb        = (float*)take((size_t)NNODE * CH * 4);   // agg, then h
    float*    cnt       = (float*)take((size_t)NNODE * 4);
    float*    stats     = (float*)take(128 * 4);
    int*      bin_count = (int*)take(64 * 4);
    int*      bin_base  = (int*)take(65 * 4);
    int*      bin_cursor= (int*)take(64 * 4);
    int*      rec_src   = (int*)take((size_t)NPAIR * 4);
    int*      rec_dst   = (int*)take((size_t)NPAIR * 4);
    float*    rec_basis = (float*)take((size_t)NPAIR * 4);
    unsigned* keys      = (unsigned*)take((size_t)NCLUST * CH * 4);

    prep_kernel<<<(NNODE * CH + 255) / 256, 256, 0, stream>>>(
        x, weight, root, xh, wh, rooth, hb, cnt, stats, bin_count, keys);
    count_kernel<<<(NEDGE + 255) / 256, 256, 0, stream>>>(ei, ea, bin_count, cnt);
    scan_kernel<<<1, 1, 0, stream>>>(bin_count, bin_base, bin_cursor);
    scatter_kernel<<<(NEDGE + 255) / 256, 256, 0, stream>>>(
        ei, ea, bin_cursor, rec_src, rec_dst, rec_basis);
    spline_gemm_kernel<<<NKERN * GPB, 32, 0, stream>>>(
        rec_src, rec_dst, rec_basis, bin_base, xh, wh, hb);
    root_elu_stats_kernel<<<(NNODE + 15) / 16, 32, 0, stream>>>(
        xh, rooth, cnt, bias, hb, stats);
    bn_pool_kernel<<<(NNODE * CH + 255) / 256, 256, 0, stream>>>(
        hb, stats, gamma, beta, pos, batch, keys);
    pool_finalize_kernel<<<(NCLUST * CH + 255) / 256, 256, 0, stream>>>(keys, out);
}
